// CrossAlignmentLayer_15298673509079
// MI455X (gfx1250) — matrix operand order repaired and checked by reading, not compiled, not run on hardware
//
#include <hip/hip_runtime.h>

// ---------------------------------------------------------------------------
// CDNA5 feature detection (compile-safe: falls back to sync copies)
// ---------------------------------------------------------------------------
#if defined(__has_builtin)
#  if __has_builtin(__builtin_amdgcn_global_load_async_to_lds_b128)
#    define HAVE_ASYNC_LDS 1
#  endif
#endif
#ifndef HAVE_ASYNC_LDS
#  define HAVE_ASYNC_LDS 0
#endif

// ---------------------------------------------------------------------------
// Types & helpers
// ---------------------------------------------------------------------------
typedef __attribute__((ext_vector_type(16))) __bf16 bf16x16;
typedef __attribute__((ext_vector_type(8)))  float  f32x8;
typedef __attribute__((ext_vector_type(4)))  int    v4i;

union FragBF { uint4 u[2]; bf16x16 v; };

__device__ __forceinline__ unsigned short f2b(float v) {
  unsigned u = __float_as_uint(v);
  u += 0x7fffu + ((u >> 16) & 1u);          // round-to-nearest-even
  return (unsigned short)(u >> 16);
}
__device__ __forceinline__ float gelu_f(float x) {
  return 0.5f * x * (1.0f + erff(x * 0.7071067811865475f));  // exact gelu
}

// 16-byte global -> LDS copy; async (ASYNCcnt) when the toolchain exposes it.
__device__ __forceinline__ void cp16_g2l(const unsigned short* g, unsigned short* l) {
#if HAVE_ASYNC_LDS
  __builtin_amdgcn_global_load_async_to_lds_b128(
      (__attribute__((address_space(1))) v4i*)(__attribute__((address_space(1))) void*)(void*)g,
      (__attribute__((address_space(3))) v4i*)(__attribute__((address_space(3))) void*)(void*)l,
      0, 0);
#else
  *(uint4*)l = *(const uint4*)g;
#endif
}
__device__ __forceinline__ void async_join() {
#if HAVE_ASYNC_LDS
#  if defined(__has_builtin) && __has_builtin(__builtin_amdgcn_s_wait_asynccnt)
  __builtin_amdgcn_s_wait_asynccnt(0);
#  else
  asm volatile("s_wait_asynccnt 0" ::: "memory");
#  endif
#endif
}

// ---------------------------------------------------------------------------
// Generic batched bf16 WMMA GEMM:  C[b,h] = A (MxK) * B (KxN), f32 accumulate.
//   transA: A element (m,k) at A[k*lda + m]   (else A[m*lda + k])
//   transB: B element (k,n) at B[n*ldb + k]   (else B[k*ldb + n])
//   K non-multiple-of-32 is zero-padded; M may be partial.
// Block tile 64(M) x 128(N), 8 waves; wave tile 16x64 (4 wmma frags).
// Double-buffered LDS; async global->LDS staging on the fast paths.
// ---------------------------------------------------------------------------
__global__ __launch_bounds__(256)
void k_gemm_bf16(const unsigned short* __restrict__ A,
                 const unsigned short* __restrict__ B,
                 float* __restrict__ C,
                 int M, int Nn, int K,
                 int lda, int ldb, int ldc,
                 long sAb, long sAh, long sBb, long sBh, long sCb, long sCh,
                 int nh, int transA, int transB)
{
  __shared__ unsigned short As[2][64][32];   // [buf][m][k]
  __shared__ unsigned short Bs[2][128][48];  // [buf][n][k] (stride 48 -> 16B rows)

  const int tid  = threadIdx.x;
  const int z    = blockIdx.z;
  const int bb   = z / nh, hh = z % nh;
  const long offA = (long)bb * sAb + (long)hh * sAh;
  const long offB = (long)bb * sBb + (long)hh * sBh;
  const long offC = (long)bb * sCb + (long)hh * sCh;
  const int m0 = blockIdx.y * 64;
  const int n0 = blockIdx.x * 128;

  const int wave = tid >> 5, lane = tid & 31;
  const int wm = wave & 3;        // 16-row slot
  const int wn = wave >> 2;       // 64-col slot
  const int half = lane >> 4, mr = lane & 15;

  // block-uniform staging-path selection
  const bool kAligned = (K & 31) == 0;
  const bool fastA  = (!transA) && kAligned && (m0 + 64 <= M);
  const bool padA   = (!transA) && kAligned && (m0 + 64 > M);   // partial-M (AV gemms)
  const bool fastBt = transB  && kAligned && (n0 + 128 <= Nn);
  const bool fastBn = (!transB) && kAligned && (n0 + 128 <= Nn);

  // per-thread staging coordinates
  const int a_row = tid >> 2;            // 0..63
  const int a_c8  = (tid & 3) << 3;      // 0,8,16,24

  auto stageA = [&](int k0, int buf) {
    const int gm = m0 + a_row;
    const int gk = k0 + a_c8;
    unsigned short* dst = &As[buf][a_row][a_c8];
    if (fastA) {
      cp16_g2l(A + offA + (long)gm * lda + gk, dst);
    } else if (padA) {
      if (gm < M) cp16_g2l(A + offA + (long)gm * lda + gk, dst);
      else        *(uint4*)dst = uint4{0, 0, 0, 0};
    } else {  // transA (K=16 logits) or ragged K
      alignas(16) unsigned short tmp[8] = {0,0,0,0,0,0,0,0};
      if (gm < M) {
        if (!transA) {
          for (int j = 0; j < 8; ++j)
            if (gk + j < K) tmp[j] = A[offA + (long)gm * lda + gk + j];
        } else {
          for (int j = 0; j < 8; ++j)
            if (gk + j < K) tmp[j] = A[offA + (long)(gk + j) * lda + gm];
        }
      }
      *(uint4*)dst = *(const uint4*)tmp;
    }
  };

  auto stageB = [&](int k0, int buf) {
    if (fastBt) {
      #pragma unroll
      for (int i = 0; i < 2; ++i) {
        const int c  = tid * 2 + i;        // 0..511
        const int n  = c >> 2;             // 0..127
        const int k8 = (c & 3) << 3;       // 0,8,16,24
        cp16_g2l(B + offB + (long)(n0 + n) * ldb + k0 + k8, &Bs[buf][n][k8]);
      }
    } else if (fastBn) {
      #pragma unroll
      for (int i = 0; i < 2; ++i) {
        const int c    = tid * 2 + i;
        const int krow = c >> 4;           // 0..31
        const int n8   = (c & 15) << 3;    // 0..120
        const long gidx = offB + (long)(k0 + krow) * ldb + n0 + n8;
        uint4 t = *(const uint4*)(B + gidx);
        if (k0 + 64 < K) __builtin_prefetch(B + gidx + 64L * ldb, 0, 1);
        const unsigned short* tp = (const unsigned short*)&t;
        #pragma unroll
        for (int j = 0; j < 8; ++j) Bs[buf][n8 + j][krow] = tp[j];  // transpose
      }
    } else {  // ragged K (K=16 logits) and/or partial N
      #pragma unroll
      for (int i = 0; i < 2; ++i) {
        const int c = tid * 2 + i;
        alignas(16) unsigned short tmp[8] = {0,0,0,0,0,0,0,0};
        if (!transB) {
          const int krow = c >> 4;
          const int n8   = (c & 15) << 3;
          const int gk = k0 + krow, gn = n0 + n8;
          if (gk < K) {
            for (int j = 0; j < 8; ++j)
              if (gn + j < Nn) tmp[j] = B[offB + (long)gk * ldb + gn + j];
          }
          for (int j = 0; j < 8; ++j) Bs[buf][n8 + j][krow] = tmp[j];
        } else {
          const int n  = c >> 2;
          const int k8 = (c & 3) << 3;
          const int gn = n0 + n, gk = k0 + k8;
          if (gn < Nn) {
            for (int j = 0; j < 8; ++j)
              if (gk + j < K) tmp[j] = B[offB + (long)gn * ldb + gk + j];
          }
          *(uint4*)&Bs[buf][n][k8] = *(const uint4*)tmp;
        }
      }
    }
  };

  f32x8 acc[4] = {};

  // prologue: stage tile 0
  stageA(0, 0);
  stageB(0, 0);
  async_join();
  __syncthreads();

  int buf = 0;
  for (int k0 = 0; k0 < K; k0 += 32) {
    // stage next tile into the other buffer while computing this one
    if (k0 + 32 < K) {
      stageA(k0 + 32, buf ^ 1);
      stageB(k0 + 32, buf ^ 1);
    }

    // ---- preload all fragments, then issue WMMAs back-to-back ----
    // A 16x32 bf16 layout: lanes 0-15 -> M rows, K {0..7,16..23};
    //                      lanes 16-31 -> same rows, K {8..15,24..31}
    // B 32x16 layout: lane n = lane%16; lanes 0-15 K {0..7,16..23}, lanes 16-31 K {8..15,24..31}
    FragBF fa, fb[4];
    const int arow = wm * 16 + mr;
    fa.u[0] = *(const uint4*)&As[buf][arow][half ? 8  : 0];
    fa.u[1] = *(const uint4*)&As[buf][arow][half ? 24 : 16];
    #pragma unroll
    for (int j = 0; j < 4; ++j) {
      const int bn = wn * 64 + j * 16 + mr;
      fb[j].u[0] = *(const uint4*)&Bs[buf][bn][half * 8];
      fb[j].u[1] = *(const uint4*)&Bs[buf][bn][16 + half * 8];
    }
    #pragma unroll
    for (int j = 0; j < 4; ++j) {
      acc[j] = __builtin_amdgcn_wmma_f32_16x16x32_bf16(
          false, fa.v, false, fb[j].v, (short)0, acc[j], false, false);
    }

    async_join();      // next buffer's async copies landed
    __syncthreads();   // publish next buffer / retire reads of this one
    buf ^= 1;
  }

  // ---- store: C/D layout: VGPR r, lanes 0-15 -> M=r,N=lane; 16-31 -> M=r+8 ----
  #pragma unroll
  for (int j = 0; j < 4; ++j) {
    const int n = n0 + wn * 64 + j * 16 + mr;
    if (n >= Nn) continue;
    #pragma unroll
    for (int r = 0; r < 8; ++r) {
      const int m = m0 + wm * 16 + r + half * 8;
      if (m < M) C[offC + (long)m * ldc + n] = acc[j][r];
    }
  }
}

// ---------------------------------------------------------------------------
// Deterministic batch statistics: stage 1 (per-block partial sum/sumsq)
// grid = (nblk, nbatch); x layout: [batch][len] contiguous
// ---------------------------------------------------------------------------
__global__ __launch_bounds__(256)
void k_stats1(const float* __restrict__ x, float* __restrict__ part, long len)
{
  const long b = blockIdx.y;
  const float* p = x + b * len;
  float s = 0.f, s2 = 0.f;
  for (long i = (long)blockIdx.x * blockDim.x + threadIdx.x; i < len;
       i += (long)gridDim.x * blockDim.x) {
    float v = p[i]; s += v; s2 += v * v;
  }
  __shared__ float sh0[256], sh1[256];
  sh0[threadIdx.x] = s; sh1[threadIdx.x] = s2;
  __syncthreads();
  for (int o = 128; o > 0; o >>= 1) {
    if (threadIdx.x < o) { sh0[threadIdx.x] += sh0[threadIdx.x + o];
                           sh1[threadIdx.x] += sh1[threadIdx.x + o]; }
    __syncthreads();
  }
  if (threadIdx.x == 0) {
    long idx = (b * gridDim.x + blockIdx.x) * 2;
    part[idx] = sh0[0]; part[idx + 1] = sh1[0];
  }
}

__global__ __launch_bounds__(256)
void k_stats2(const float* __restrict__ part, float* __restrict__ stats, int nparts)
{
  const int b = blockIdx.x;
  float s = 0.f, s2 = 0.f;
  for (int i = threadIdx.x; i < nparts; i += blockDim.x) {
    s  += part[((long)b * nparts + i) * 2];
    s2 += part[((long)b * nparts + i) * 2 + 1];
  }
  __shared__ float sh0[256], sh1[256];
  sh0[threadIdx.x] = s; sh1[threadIdx.x] = s2;
  __syncthreads();
  for (int o = 128; o > 0; o >>= 1) {
    if (threadIdx.x < o) { sh0[threadIdx.x] += sh0[threadIdx.x + o];
                           sh1[threadIdx.x] += sh1[threadIdx.x + o]; }
    __syncthreads();
  }
  if (threadIdx.x == 0) { stats[b * 2] = sh0[0]; stats[b * 2 + 1] = sh1[0]; }
}

// ---------------------------------------------------------------------------
// LayerNorm (batch-wide stats) + optional affine + optional gelu,
// writes f32 and/or bf16 outputs.
// ---------------------------------------------------------------------------
__global__ __launch_bounds__(256)
void k_ln_act(const float* __restrict__ x, const float* __restrict__ stats,
              const float* __restrict__ g, const float* __restrict__ bia,
              float* __restrict__ of, unsigned short* __restrict__ ob,
              long len, int Ld, int act, long total)
{
  long i = (long)blockIdx.x * blockDim.x + threadIdx.x;
  if (i >= total) return;
  const long b = i / len;
  const long r = i - b * len;
  const int  c = (int)(r / Ld);
  const float mu  = stats[b * 2] / (float)len;
  const float var = stats[b * 2 + 1] / (float)len - mu * mu;
  float y = (x[i] - mu) * rsqrtf(var + 1e-5f);
  if (g) y = y * g[c] + bia[c];
  if (act) y = gelu_f(y);
  if (of) of[i] = y;
  if (ob) ob[i] = f2b(y);
}

// ---------------------------------------------------------------------------
// Row softmax over batch-normalized logits -> bf16 weights.
// ---------------------------------------------------------------------------
__global__ __launch_bounds__(256)
void k_softmax(const float* __restrict__ logits, const float* __restrict__ stats,
               unsigned short* __restrict__ out, int rowlen, int rows_per_batch,
               long meanlen)
{
  const long row = blockIdx.x;
  const long b   = row / rows_per_batch;
  const float* p = logits + row * (long)rowlen;
  unsigned short* o = out + row * (long)rowlen;
  const float mu  = stats[b * 2] / (float)meanlen;
  const float var = stats[b * 2 + 1] / (float)meanlen - mu * mu;
  const float inv = rsqrtf(var + 1e-5f);
  const int tid = threadIdx.x;
  __shared__ float sh[256];

  float mx = -3.0e38f;
  for (int i = tid; i < rowlen; i += 256) mx = fmaxf(mx, (p[i] - mu) * inv);
  sh[tid] = mx; __syncthreads();
  for (int oo = 128; oo > 0; oo >>= 1) {
    if (tid < oo) sh[tid] = fmaxf(sh[tid], sh[tid + oo]);
    __syncthreads();
  }
  const float rmax = sh[0]; __syncthreads();

  float sum = 0.f;
  for (int i = tid; i < rowlen; i += 256) sum += expf((p[i] - mu) * inv - rmax);
  sh[tid] = sum; __syncthreads();
  for (int oo = 128; oo > 0; oo >>= 1) {
    if (tid < oo) sh[tid] += sh[tid + oo];
    __syncthreads();
  }
  const float rs = 1.f / sh[0]; __syncthreads();

  for (int i = tid; i < rowlen; i += 256)
    o[i] = f2b(expf((p[i] - mu) * inv - rmax) * rs);
}

// ---------------------------------------------------------------------------
// Pointwise kernels
// ---------------------------------------------------------------------------
__global__ __launch_bounds__(256)
void k_cvt_bf16(const float* __restrict__ x, unsigned short* __restrict__ y, long n)
{
  long i = (long)blockIdx.x * blockDim.x + threadIdx.x;
  if (i < n) y[i] = f2b(x[i]);
}
__global__ __launch_bounds__(256)
void k_add(const float* __restrict__ a, const float* __restrict__ b,
           float* __restrict__ o, long n)
{
  long i = (long)blockIdx.x * blockDim.x + threadIdx.x;
  if (i < n) o[i] = a[i] + b[i];
}
__global__ __launch_bounds__(256)
void k_gelu_bf(const float* __restrict__ a, unsigned short* __restrict__ ob, long n)
{
  long i = (long)blockIdx.x * blockDim.x + threadIdx.x;
  if (i < n) ob[i] = f2b(gelu_f(a[i]));
}
__global__ __launch_bounds__(256)
void k_add_gelu(const float* __restrict__ a, const float* __restrict__ b,
                float* __restrict__ of, unsigned short* __restrict__ ob, long n)
{
  long i = (long)blockIdx.x * blockDim.x + threadIdx.x;
  if (i >= n) return;
  float v = gelu_f(a[i] + b[i]);
  of[i] = v;
  if (ob) ob[i] = f2b(v);
}

// ---------------------------------------------------------------------------
// Host orchestration
// ---------------------------------------------------------------------------
extern "C" void kernel_launch(void* const* d_in, const int* in_sizes, int n_in,
                              void* d_out, int out_size, void* d_ws, size_t ws_size,
                              hipStream_t stream)
{
  (void)in_sizes; (void)n_in; (void)out_size; (void)ws_size;

  // ---- inputs (setup_inputs dict order, params flattened in insertion order)
  const float* input_f = (const float*)d_in[0];
  const float* query_f = (const float*)d_in[1];
  const float* Wi  = (const float*)d_in[2];  const float* gi  = (const float*)d_in[3];  const float* bi_  = (const float*)d_in[4];
  const float* Wq  = (const float*)d_in[5];  const float* gq  = (const float*)d_in[6];  const float* bq_  = (const float*)d_in[7];
  const float* Wqq = (const float*)d_in[8];  const float* gqq = (const float*)d_in[9];  const float* bqq  = (const float*)d_in[10];
  const float* Wkv = (const float*)d_in[11]; const float* gkv = (const float*)d_in[12]; const float* bkv  = (const float*)d_in[13];
  const float* W3c = (const float*)d_in[14]; const float* g3c = (const float*)d_in[15]; const float* b3c  = (const float*)d_in[16];
  const float* Wqkv= (const float*)d_in[17]; const float* gqkv= (const float*)d_in[18]; const float* bqkv = (const float*)d_in[19];
  const float* W3  = (const float*)d_in[20]; const float* g3  = (const float*)d_in[21]; const float* b3_  = (const float*)d_in[22];
  const float* Wf1 = (const float*)d_in[23]; const float* gf1 = (const float*)d_in[24]; const float* bf1  = (const float*)d_in[25];
  const float* Wf2 = (const float*)d_in[26]; const float* gf2 = (const float*)d_in[27]; const float* bf2  = (const float*)d_in[28];

  // ---- dims
  const long E_in = 8L * 2048 * 4096;
  const long E_q  = 8L * 256 * 256;
  const long Y1e  = 8L * 256 * 4096;
  const long Ykve = 8L * 384 * 4096;
  const long Lge  = 8L * 8 * 256 * 4096;
  const long Qkve = 8L * 512 * 256;

  // ---- workspace carve
  char* W = (char*)d_ws;
  size_t off = 0;
  auto carve = [&](size_t bytes) { size_t r = off; off += (bytes + 255) & ~(size_t)255; return r; };

  const size_t inbf_b = carve(E_in * 2);        // bf16 input_f; reused as cross softmax weights
  const size_t qfbf_b = carve(E_q * 2);
  const size_t wib  = carve(256L*2048*2); const size_t wqb  = carve(256L*256*2);
  const size_t wqqb = carve(128L*256*2);  const size_t wkvb = carve(384L*256*2);
  const size_t w3cb = carve(256L*256*2);  const size_t wqkvb= carve(512L*256*2);
  const size_t w3b  = carve(256L*256*2);  const size_t wf1b = carve(512L*256*2);
  const size_t wf2b = carve(256L*512*2);
  const size_t Y1_b  = carve(Y1e * 4);
  const size_t isp_b = carve(Y1e * 2);
  const size_t Yq_b  = carve(E_q * 4);
  const size_t qsp_b = carve(E_q * 2);
  const size_t Yqq_b = carve(8L*128*256*4);
  const size_t qq_b  = carve(8L*128*256*2);
  const size_t Ykv_b = carve(Ykve * 4);
  const size_t kvb_b = carve(Ykve * 2);
  const size_t lg_b  = carve(Lge * 4);          // 268MB; reused for self-attn logits/weights
  const size_t r1_b  = carve(E_q * 4);
  const size_t r1b_b = carve(E_q * 2);
  const size_t Y3c_b = carve(E_q * 4);
  const size_t crs_b = carve(E_q * 4);
  const size_t qf_b  = carve(E_q * 4);
  const size_t qfg_b = carve(E_q * 2);
  const size_t Yqkv_b= carve(Qkve * 4);
  const size_t qkvb_b= carve(Qkve * 2);
  const size_t lg2_b = lg_b;                         // 16MB alias inside logits region
  const size_t w2b_b = lg_b + (64UL << 20);          // bf16 weights, disjoint in region
  const size_t r2_b  = carve(E_q * 4);
  const size_t r2b_b = carve(E_q * 2);
  const size_t Y3_b  = carve(E_q * 4);
  const size_t sa_b  = carve(E_q * 4);
  const size_t qf2_b = carve(E_q * 4);
  const size_t qf2b_b= carve(E_q * 2);
  const size_t Yf1_b = carve(Qkve * 4);
  const size_t f1b_b = carve(Qkve * 2);
  const size_t Yf2_b = carve(E_q * 4);
  const size_t ffn_b = carve(E_q * 4);
  const size_t part_b= carve(8L * 256 * 2 * 4);
  const size_t stat_b= carve(64L * 2 * 4);

  float* partP = (float*)(W + part_b);
  float* statP = (float*)(W + stat_b);

  // ---- launch helpers
  auto cvt = [&](const float* x, size_t ob, long n) {
    k_cvt_bf16<<<(int)((n + 255) / 256), 256, 0, stream>>>(x, (unsigned short*)(W + ob), n);
  };
  auto gemm = [&](size_t Ab, size_t Bb, size_t Cb, int M, int Nn, int K,
                  int lda, int ldb, int ldc,
                  long sAb, long sAh, long sBb, long sBh, long sCb, long sCh,
                  int nb, int nh, int tA, int tB) {
    dim3 g((Nn + 127) / 128, (M + 63) / 64, nb * nh);
    k_gemm_bf16<<<g, 256, 0, stream>>>((const unsigned short*)(W + Ab),
                                       (const unsigned short*)(W + Bb),
                                       (float*)(W + Cb),
                                       M, Nn, K, lda, ldb, ldc,
                                       sAb, sAh, sBb, sBh, sCb, sCh, nh, tA, tB);
  };
  auto stats = [&](size_t Xb, int slot, long len, int nb, int nblk) {
    k_stats1<<<dim3(nblk, nb), 256, 0, stream>>>((const float*)(W + Xb), partP, len);
    k_stats2<<<nb, 256, 0, stream>>>(partP, statP + slot * 2, nblk);
  };
  auto lnact = [&](size_t Xb, int slot, const float* g_, const float* b_,
                   void* of, void* ob, long len, int Ld, int act, int nb) {
    long total = len * nb;
    k_ln_act<<<(int)((total + 255) / 256), 256, 0, stream>>>(
        (const float*)(W + Xb), statP + slot * 2, g_, b_,
        (float*)of, (unsigned short*)ob, len, Ld, act, total);
  };

  // ====================== pipeline ======================
  cvt(input_f, inbf_b, E_in);
  cvt(query_f, qfbf_b, E_q);
  cvt(Wi,  wib,  256L*2048); cvt(Wq,  wqb,  256L*256); cvt(Wqq, wqqb, 128L*256);
  cvt(Wkv, wkvb, 384L*256);  cvt(W3c, w3cb, 256L*256); cvt(Wqkv,wqkvb,512L*256);
  cvt(W3,  w3b,  256L*256);  cvt(Wf1, wf1b, 512L*256); cvt(Wf2, wf2b, 256L*512);

  // input_space = conv_ln(input_f, Wi, act) [8,256,4096]
  gemm(wib, inbf_b, Y1_b, 256, 4096, 2048, 2048, 4096, 4096,
       0, 0, 2048L*4096, 0, 256L*4096, 0, 8, 1, 0, 0);
  stats(Y1_b, 0, 256L*4096, 8, 256);
  lnact(Y1_b, 0, gi, bi_, nullptr, W + isp_b, 256L*4096, 4096, 1, 8);

  // query_space = conv_ln(query_f, Wq, act) [8,256,256]
  gemm(wqb, qfbf_b, Yq_b, 256, 256, 256, 256, 256, 256,
       0, 0, 65536, 0, 65536, 0, 8, 1, 0, 0);
  stats(Yq_b, 1, 65536, 8, 64);
  lnact(Yq_b, 1, gq, bq_, nullptr, W + qsp_b, 65536, 256, 1, 8);

  // query_q = conv_ln(query_space, Wqq) [8,128,256]
  gemm(wqqb, qsp_b, Yqq_b, 128, 256, 256, 256, 256, 256,
       0, 0, 65536, 0, 32768, 0, 8, 1, 0, 0);
  stats(Yqq_b, 2, 32768, 8, 32);
  lnact(Yqq_b, 2, gqq, bqq, nullptr, W + qq_b, 32768, 256, 0, 8);

  // input_kv = conv_ln(input_space, Wkv) [8,384,4096]
  gemm(wkvb, isp_b, Ykv_b, 384, 4096, 256, 256, 4096, 4096,
       0, 0, 256L*4096, 0, 384L*4096, 0, 8, 1, 0, 0);
  stats(Ykv_b, 3, 384L*4096, 8, 256);
  lnact(Ykv_b, 3, gkv, bkv, nullptr, W + kvb_b, 384L*4096, 4096, 0, 8);

  // cross logits[b,h,l,m] = q^T k   (K=16 zero-padded to 32)
  gemm(qq_b, kvb_b, lg_b, 256, 4096, 16, 256, 4096, 4096,
       32768, 4096, 384L*4096, 16L*4096, 8L*256*4096, 256L*4096, 8, 8, 1, 0);
  stats(lg_b, 4, 8L*256*4096, 8, 256);
  k_softmax<<<16384, 256, 0, stream>>>((const float*)(W + lg_b), statP + 4 * 2,
                                       (unsigned short*)(W + inbf_b), 4096, 2048,
                                       8L*256*4096);
  // r1[b, h*32+d, l] = sum_m w[l,m] v[d,m]   (A=v, B=w^T via transB)
  gemm(kvb_b + 128L*4096*2, inbf_b, r1_b, 32, 256, 4096, 4096, 4096, 256,
       384L*4096, 32L*4096, 8L*256*4096, 256L*4096, 65536, 8192, 8, 8, 0, 1);
  stats(r1_b, 5, 65536, 8, 64);
  lnact(r1_b, 5, nullptr, nullptr, nullptr, W + r1b_b, 65536, 256, 1, 8);

  // cross = conv_ln(r1, W3c)
  gemm(w3cb, r1b_b, Y3c_b, 256, 256, 256, 256, 256, 256,
       0, 0, 65536, 0, 65536, 0, 8, 1, 0, 0);
  stats(Y3c_b, 6, 65536, 8, 64);
  lnact(Y3c_b, 6, g3c, b3c, W + crs_b, nullptr, 65536, 256, 0, 8);

  // qf = query_f + cross ; gelu(qf) -> bf16
  k_add<<<(int)((E_q + 255) / 256), 256, 0, stream>>>(query_f, (const float*)(W + crs_b),
                                                      (float*)(W + qf_b), E_q);
  k_gelu_bf<<<(int)((E_q + 255) / 256), 256, 0, stream>>>((const float*)(W + qf_b),
                                                          (unsigned short*)(W + qfg_b), E_q);

  // qkv = conv_ln(gelu(qf), Wqkv) [8,512,256]
  gemm(wqkvb, qfg_b, Yqkv_b, 512, 256, 256, 256, 256, 256,
       0, 0, 65536, 0, 131072, 0, 8, 1, 0, 0);
  stats(Yqkv_b, 7, 131072, 8, 128);
  lnact(Yqkv_b, 7, gqkv, bqkv, nullptr, W + qkvb_b, 131072, 256, 0, 8);

  // self logits[b,h,l,m] = sq^T sk  (K=16 padded)
  gemm(qkvb_b, qkvb_b + 128L*256*2, lg2_b, 256, 256, 16, 256, 256, 256,
       131072, 4096, 131072, 4096, 8L*65536, 65536, 8, 8, 1, 0);
  stats(lg2_b, 8, 8L*65536, 8, 128);
  k_softmax<<<16384, 256, 0, stream>>>((const float*)(W + lg2_b), statP + 8 * 2,
                                       (unsigned short*)(W + w2b_b), 256, 2048,
                                       8L*65536);
  // r2 = w2 @ sv^T
  gemm(qkvb_b + 256L*256*2, w2b_b, r2_b, 32, 256, 256, 256, 256, 256,
       131072, 8192, 8L*65536, 65536, 65536, 8192, 8, 8, 0, 1);
  stats(r2_b, 9, 65536, 8, 64);
  lnact(r2_b, 9, nullptr, nullptr, nullptr, W + r2b_b, 65536, 256, 1, 8);

  // sa = conv_ln(r2, W3)
  gemm(w3b, r2b_b, Y3_b, 256, 256, 256, 256, 256, 256,
       0, 0, 65536, 0, 65536, 0, 8, 1, 0, 0);
  stats(Y3_b, 10, 65536, 8, 64);
  lnact(Y3_b, 10, g3, b3_, W + sa_b, nullptr, 65536, 256, 0, 8);

  // qf2 = gelu(qf + sa)  (f32 + bf16)
  k_add_gelu<<<(int)((E_q + 255) / 256), 256, 0, stream>>>(
      (const float*)(W + qf_b), (const float*)(W + sa_b),
      (float*)(W + qf2_b), (unsigned short*)(W + qf2b_b), E_q);

  // ffn = conv_ln(qf2, Wf1, act) -> conv_ln(., Wf2)
  gemm(wf1b, qf2b_b, Yf1_b, 512, 256, 256, 256, 256, 256,
       0, 0, 65536, 0, 131072, 0, 8, 1, 0, 0);
  stats(Yf1_b, 11, 131072, 8, 128);
  lnact(Yf1_b, 11, gf1, bf1, nullptr, W + f1b_b, 131072, 256, 1, 8);

  gemm(wf2b, f1b_b, Yf2_b, 256, 256, 512, 512, 256, 256,
       0, 0, 131072, 0, 65536, 0, 8, 1, 0, 0);
  stats(Yf2_b, 12, 65536, 8, 64);
  lnact(Yf2_b, 12, gf2, bf2, W + ffn_b, nullptr, 65536, 256, 0, 8);

  // out = gelu(qf2 + ffn)
  k_add_gelu<<<(int)((E_q + 255) / 256), 256, 0, stream>>>(
      (const float*)(W + qf2_b), (const float*)(W + ffn_b),
      (float*)d_out, nullptr, E_q);
}
